// GTN_27908697489426
// MI455X (gfx1250) — matrix operand order repaired and checked by reading, not compiled, not run on hardware
//
#include <hip/hip_runtime.h>
#include <hip/hip_bf16.h>

// ---------- problem constants ----------
constexpr int E_  = 4;
constexpr int C_  = 2;
constexpr int NN  = 2048;   // nodes
constexpr int DIN_ = 256;
constexpr int DOUT_ = 128;
constexpr int P_  = 8192;   // pairs
constexpr int MLP_IN_ = 512;

// ---------- WMMA types / helpers ----------
typedef __attribute__((ext_vector_type(16))) _Float16 v16h;
typedef __attribute__((ext_vector_type(8)))  float    v8f;

union FragH { v16h v; uint4 q[2]; };

#define WMMA_F16(a, b, c) \
  __builtin_amdgcn_wmma_f32_16x16x32_f16(false, (a), false, (b), (short)0, (c), false, false)

// A-matrix fragment (16x32 f16), row-major source [.., lda], per ISA 7.12.2:
// lanes 0-15: row M=l, K 0..7 then 16..23; lanes 16-31: row M=l-16, K 8..15 then 24..31.
__device__ __forceinline__ v16h load_a_frag(const _Float16* __restrict__ A, int lda,
                                            int row_base, int k0, int lane) {
  int row  = row_base + (lane & 15);
  int koff = (lane & 16) ? 8 : 0;
  const _Float16* p = A + (size_t)row * lda + k0 + koff;
  FragH f;
  f.q[0] = *(const uint4*)(p);
  f.q[1] = *(const uint4*)(p + 16);
  return f.v;
}

// B-matrix fragment (32x16 f16) from B^T storage [N, K] row-major:
// lanes 0-15: col N=l, K 0..7 then 16..23; lanes 16-31: col N=l-16, K 8..15 then 24..31.
__device__ __forceinline__ v16h load_bt_frag(const _Float16* __restrict__ Bt, int ldb,
                                             int col_base, int k0, int lane) {
  int col = col_base + (lane & 15);
  int kb  = k0 + ((lane & 16) ? 8 : 0);
  const uint4* p = (const uint4*)(Bt + (size_t)col * ldb + kb);
  FragH f;
  f.q[0] = p[0];
  f.q[1] = p[2];
  return f.v;
}

// ---------- K0a: zero deg + loss ----------
__global__ void k_init(float* __restrict__ deg, float* __restrict__ loss) {
  int i = blockIdx.x * blockDim.x + threadIdx.x;
  if (i < 2 * C_ * NN) deg[i] = 0.0f;
  if (i == 0) *loss = 0.0f;
}

// ---------- K0b: cast+transpose MLP weights to f16 ----------
__global__ void k_prep_w(const float* __restrict__ Wm1, const float* __restrict__ Wm2,
                         _Float16* __restrict__ Wm1hT, _Float16* __restrict__ Wm2hT) {
  int i = blockIdx.x * blockDim.x + threadIdx.x;
  if (i < MLP_IN_ * 256) {           // Wm1 [512,256] -> Wm1hT [256,512]
    int k = i >> 8, j = i & 255;
    Wm1hT[(size_t)j * MLP_IN_ + k] = (_Float16)Wm1[i];
  }
  if (i < 256 * 128) {               // Wm2 [256,128] -> Wm2hT [128,256]
    int k = i >> 7, j = i & 127;
    Wm2hT[(size_t)j * 256 + k] = (_Float16)Wm2[i];
  }
}

// ---------- K1: softmax(W)·A -> A1 (f16, row-major) and A2^T (f16, via LDS transpose) ----------
__global__ void k_mix_transpose(const float* __restrict__ A, const float* __restrict__ W1,
                                const float* __restrict__ W2,
                                _Float16* __restrict__ A1h, _Float16* __restrict__ A2hT) {
  __shared__ _Float16 tile[C_][32][33];
  int tx = threadIdx.x, ty = threadIdx.y;
  int n0 = blockIdx.y * 32, k0 = blockIdx.x * 32;

  float f1[C_][E_], f2[C_][E_];
#pragma unroll
  for (int c = 0; c < C_; ++c) {
    float m1 = -1e30f, m2 = -1e30f;
#pragma unroll
    for (int e = 0; e < E_; ++e) { m1 = fmaxf(m1, W1[c*E_+e]); m2 = fmaxf(m2, W2[c*E_+e]); }
    float s1 = 0.f, s2 = 0.f;
#pragma unroll
    for (int e = 0; e < E_; ++e) {
      f1[c][e] = __expf(W1[c*E_+e] - m1); s1 += f1[c][e];
      f2[c][e] = __expf(W2[c*E_+e] - m2); s2 += f2[c][e];
    }
    float r1 = 1.f / s1, r2 = 1.f / s2;
#pragma unroll
    for (int e = 0; e < E_; ++e) { f1[c][e] *= r1; f2[c][e] *= r2; }
  }

#pragma unroll
  for (int j = 0; j < 4; ++j) {
    int rl = ty + 8 * j;
    int n = n0 + rl, k = k0 + tx;
    float av[E_];
#pragma unroll
    for (int e = 0; e < E_; ++e) av[e] = A[((size_t)e * NN + n) * NN + k];
#pragma unroll
    for (int c = 0; c < C_; ++c) {
      float a1 = 0.f, a2 = 0.f;
#pragma unroll
      for (int e = 0; e < E_; ++e) { a1 += f1[c][e] * av[e]; a2 += f2[c][e] * av[e]; }
      A1h[((size_t)c * NN + n) * NN + k] = (_Float16)a1;
      tile[c][tx][rl] = (_Float16)a2;          // [col_local][row_local]
    }
  }
  __syncthreads();
#pragma unroll
  for (int j = 0; j < 4; ++j) {
    int rl = ty + 8 * j;
    int kk = k0 + rl, nnn = n0 + tx;
#pragma unroll
    for (int c = 0; c < C_; ++c)
      A2hT[((size_t)c * NN + kk) * NN + nnn] = tile[c][rl][tx];
  }
}

// ---------- K2: H = A1 @ A2 (per channel), diag-zero fused, store H^T (f16), deg column sums ----------
__global__ void k_spspmm(const _Float16* __restrict__ A1h, const _Float16* __restrict__ A2hT,
                         _Float16* __restrict__ Ht, float* __restrict__ deg) {
  int lane = threadIdx.x & 31, wid = threadIdx.x >> 5;
  int c = blockIdx.y;
  int tilei = blockIdx.x * 4 + wid;          // 0..4095
  int m0 = (tilei >> 6) * 32;                // H row block
  int n0 = (tilei & 63) * 32;                // H col block
  const _Float16* Ac = A1h + (size_t)c * NN * NN;
  const _Float16* Bc = A2hT + (size_t)c * NN * NN;
  _Float16* Hc = Ht + (size_t)c * NN * NN;
  float* degc = deg + c * NN;

  v8f acc00 = {}, acc01 = {}, acc10 = {}, acc11 = {};
  for (int k0 = 0; k0 < NN; k0 += 32) {
    __builtin_prefetch(Ac + (size_t)(m0 + (lane & 15)) * NN + k0 + 64, 0, 3);
    __builtin_prefetch(Bc + (size_t)(n0 + (lane & 15)) * NN + k0 + 64, 0, 3);
    v16h a0 = load_a_frag(Ac, NN, m0,      k0, lane);
    v16h a1 = load_a_frag(Ac, NN, m0 + 16, k0, lane);
    v16h b0 = load_bt_frag(Bc, NN, n0,      k0, lane);
    v16h b1 = load_bt_frag(Bc, NN, n0 + 16, k0, lane);
    acc00 = WMMA_F16(a0, b0, acc00);
    acc01 = WMMA_F16(a0, b1, acc01);
    acc10 = WMMA_F16(a1, b0, acc10);
    acc11 = WMMA_F16(a1, b1, acc11);
  }

  int nlo = lane & 15;
  int mofs = (lane & 16) ? 8 : 0;
  {
    int Ni = n0 + nlo;   // H column
    float cs = 0.f;
#pragma unroll
    for (int r = 0; r < 8; ++r) {
      int Mi = m0 + r + mofs;
      float v = acc00[r]; if (Mi == Ni) v = 0.f;
      Hc[(size_t)Ni * NN + Mi] = (_Float16)v; cs += v;
      int Mi2 = m0 + 16 + r + mofs;
      float v2 = acc10[r]; if (Mi2 == Ni) v2 = 0.f;
      Hc[(size_t)Ni * NN + Mi2] = (_Float16)v2; cs += v2;
    }
    cs += __shfl_xor(cs, 16, 32);
    if (lane < 16) atomicAdd(&degc[Ni], cs);
  }
  {
    int Ni = n0 + 16 + nlo;
    float cs = 0.f;
#pragma unroll
    for (int r = 0; r < 8; ++r) {
      int Mi = m0 + r + mofs;
      float v = acc01[r]; if (Mi == Ni) v = 0.f;
      Hc[(size_t)Ni * NN + Mi] = (_Float16)v; cs += v;
      int Mi2 = m0 + 16 + r + mofs;
      float v2 = acc11[r]; if (Mi2 == Ni) v2 = 0.f;
      Hc[(size_t)Ni * NN + Mi2] = (_Float16)v2; cs += v2;
    }
    cs += __shfl_xor(cs, 16, 32);
    if (lane < 16) atomicAdd(&degc[Ni], cs);
  }
}

// ---------- K3: XL = X @ Wg, stored transposed f16 [DOUT, N] ----------
__global__ void k_xl(const float* __restrict__ X, const float* __restrict__ Wg,
                     _Float16* __restrict__ XLT) {
  int idx = blockIdx.x * blockDim.x + threadIdx.x;   // N * DOUT
  int n = idx >> 7, d = idx & 127;
  float acc = 0.f;
  for (int k = 0; k < DIN_; ++k) acc += X[(size_t)n * DIN_ + k] * Wg[(size_t)k * DOUT_ + d];
  XLT[(size_t)d * NN + n] = (_Float16)acc;
}

// ---------- K4: out = relu(dinv[m] * (H^T @ XL) + bg); write f32 to d_out + f16 copy ----------
__global__ void k_gcn_out(const _Float16* __restrict__ Ht, const _Float16* __restrict__ XLT,
                          const float* __restrict__ deg, const float* __restrict__ bg,
                          float* __restrict__ Xout, _Float16* __restrict__ Xh) {
  int lane = threadIdx.x & 31, wid = threadIdx.x >> 5;
  int c = blockIdx.y;
  int tilei = blockIdx.x * 4 + wid;          // 0..255
  int m0 = (tilei >> 2) * 32;                // node index block
  int n0 = (tilei & 3) * 32;                 // feature block
  const _Float16* Ac = Ht + (size_t)c * NN * NN;
  const float* degc = deg + c * NN;

  v8f acc00 = {}, acc01 = {}, acc10 = {}, acc11 = {};
  for (int k0 = 0; k0 < NN; k0 += 32) {
    v16h a0 = load_a_frag(Ac, NN, m0,      k0, lane);
    v16h a1 = load_a_frag(Ac, NN, m0 + 16, k0, lane);
    v16h b0 = load_bt_frag(XLT, NN, n0,      k0, lane);
    v16h b1 = load_bt_frag(XLT, NN, n0 + 16, k0, lane);
    acc00 = WMMA_F16(a0, b0, acc00);
    acc01 = WMMA_F16(a0, b1, acc01);
    acc10 = WMMA_F16(a1, b0, acc10);
    acc11 = WMMA_F16(a1, b1, acc11);
  }

  int nlo = lane & 15;
  int mofs = (lane & 16) ? 8 : 0;
  float bg0 = bg[n0 + nlo], bg1 = bg[n0 + 16 + nlo];
#pragma unroll
  for (int r = 0; r < 8; ++r) {
    int Mi = m0 + r + mofs;
    int Mi2 = Mi + 16;
    float dg = degc[Mi];   float di  = (dg  == 0.f) ? 0.f : 1.f / dg;
    float dg2 = degc[Mi2]; float di2 = (dg2 == 0.f) ? 0.f : 1.f / dg2;
    float v00 = fmaxf(di  * acc00[r] + bg0, 0.f);
    float v01 = fmaxf(di  * acc01[r] + bg1, 0.f);
    float v10 = fmaxf(di2 * acc10[r] + bg0, 0.f);
    float v11 = fmaxf(di2 * acc11[r] + bg1, 0.f);
    size_t o0 = (size_t)Mi  * (C_ * DOUT_) + c * DOUT_ + n0 + nlo;
    size_t o1 = (size_t)Mi2 * (C_ * DOUT_) + c * DOUT_ + n0 + nlo;
    Xout[o0]      = v00; Xh[o0]      = (_Float16)v00;
    Xout[o0 + 16] = v01; Xh[o0 + 16] = (_Float16)v01;
    Xout[o1]      = v10; Xh[o1]      = (_Float16)v10;
    Xout[o1 + 16] = v11; Xh[o1 + 16] = (_Float16)v11;
  }
}

// ---------- K5: h1 = relu(concat(Xu[u], Xv[v]) @ Wm1 + bm1), gathered A rows ----------
__global__ void k_mlp1(const _Float16* __restrict__ Xuh, const _Float16* __restrict__ Xvh,
                       const int* __restrict__ u_idx, const int* __restrict__ v_idx,
                       const _Float16* __restrict__ Wm1hT, const float* __restrict__ bm1,
                       _Float16* __restrict__ h1h) {
  int lane = threadIdx.x & 31, wid = threadIdx.x >> 5;
  int tilei = blockIdx.x * 4 + wid;          // 0..2047
  int m0 = (tilei >> 3) * 32;
  int n0 = (tilei & 7) * 32;

  int r0 = m0 + (lane & 15);
  int r1 = r0 + 16;
  const _Float16* ru0 = Xuh + (size_t)u_idx[r0] * (C_ * DOUT_);
  const _Float16* rv0 = Xvh + (size_t)v_idx[r0] * (C_ * DOUT_);
  const _Float16* ru1 = Xuh + (size_t)u_idx[r1] * (C_ * DOUT_);
  const _Float16* rv1 = Xvh + (size_t)v_idx[r1] * (C_ * DOUT_);
  int koff = (lane & 16) ? 8 : 0;

  v8f acc00 = {}, acc01 = {}, acc10 = {}, acc11 = {};
  for (int k0 = 0; k0 < MLP_IN_; k0 += 32) {
    const _Float16* s0 = (k0 < 256) ? (ru0 + k0) : (rv0 + (k0 - 256));
    const _Float16* s1 = (k0 < 256) ? (ru1 + k0) : (rv1 + (k0 - 256));
    FragH fa0, fa1;
    fa0.q[0] = *(const uint4*)(s0 + koff); fa0.q[1] = *(const uint4*)(s0 + koff + 16);
    fa1.q[0] = *(const uint4*)(s1 + koff); fa1.q[1] = *(const uint4*)(s1 + koff + 16);
    v16h b0 = load_bt_frag(Wm1hT, MLP_IN_, n0,      k0, lane);
    v16h b1 = load_bt_frag(Wm1hT, MLP_IN_, n0 + 16, k0, lane);
    acc00 = WMMA_F16(fa0.v, b0, acc00);
    acc01 = WMMA_F16(fa0.v, b1, acc01);
    acc10 = WMMA_F16(fa1.v, b0, acc10);
    acc11 = WMMA_F16(fa1.v, b1, acc11);
  }

  int nlo = lane & 15;
  int mofs = (lane & 16) ? 8 : 0;
  float b0v = bm1[n0 + nlo], b1v = bm1[n0 + 16 + nlo];
#pragma unroll
  for (int r = 0; r < 8; ++r) {
    int Mi = m0 + r + mofs, Mi2 = Mi + 16;
    h1h[(size_t)Mi  * 256 + n0 + nlo]      = (_Float16)fmaxf(acc00[r] + b0v, 0.f);
    h1h[(size_t)Mi  * 256 + n0 + nlo + 16] = (_Float16)fmaxf(acc01[r] + b1v, 0.f);
    h1h[(size_t)Mi2 * 256 + n0 + nlo]      = (_Float16)fmaxf(acc10[r] + b0v, 0.f);
    h1h[(size_t)Mi2 * 256 + n0 + nlo + 16] = (_Float16)fmaxf(acc11[r] + b1v, 0.f);
  }
}

// ---------- K6: h2 = relu(h1 @ Wm2 + bm2) ----------
__global__ void k_mlp2(const _Float16* __restrict__ h1h, const _Float16* __restrict__ Wm2hT,
                       const float* __restrict__ bm2, _Float16* __restrict__ h2h) {
  int lane = threadIdx.x & 31, wid = threadIdx.x >> 5;
  int tilei = blockIdx.x * 4 + wid;          // 0..1023
  int m0 = (tilei >> 2) * 32;
  int n0 = (tilei & 3) * 32;

  v8f acc00 = {}, acc01 = {}, acc10 = {}, acc11 = {};
  for (int k0 = 0; k0 < 256; k0 += 32) {
    v16h a0 = load_a_frag(h1h, 256, m0,      k0, lane);
    v16h a1 = load_a_frag(h1h, 256, m0 + 16, k0, lane);
    v16h b0 = load_bt_frag(Wm2hT, 256, n0,      k0, lane);
    v16h b1 = load_bt_frag(Wm2hT, 256, n0 + 16, k0, lane);
    acc00 = WMMA_F16(a0, b0, acc00);
    acc01 = WMMA_F16(a0, b1, acc01);
    acc10 = WMMA_F16(a1, b0, acc10);
    acc11 = WMMA_F16(a1, b1, acc11);
  }

  int nlo = lane & 15;
  int mofs = (lane & 16) ? 8 : 0;
  float b0v = bm2[n0 + nlo], b1v = bm2[n0 + 16 + nlo];
#pragma unroll
  for (int r = 0; r < 8; ++r) {
    int Mi = m0 + r + mofs, Mi2 = Mi + 16;
    h2h[(size_t)Mi  * 128 + n0 + nlo]      = (_Float16)fmaxf(acc00[r] + b0v, 0.f);
    h2h[(size_t)Mi  * 128 + n0 + nlo + 16] = (_Float16)fmaxf(acc01[r] + b1v, 0.f);
    h2h[(size_t)Mi2 * 128 + n0 + nlo]      = (_Float16)fmaxf(acc10[r] + b0v, 0.f);
    h2h[(size_t)Mi2 * 128 + n0 + nlo + 16] = (_Float16)fmaxf(acc11[r] + b1v, 0.f);
  }
}

// ---------- K7: logits -> softmax B -> double-softmax CE loss ----------
__global__ void k_head(const _Float16* __restrict__ h2h, const float* __restrict__ Wm3,
                       const float* __restrict__ bm3, const int* __restrict__ labels,
                       float* __restrict__ Bout, float* __restrict__ loss) {
  __shared__ float w[256];
  __shared__ float b3[2];
  int tid = threadIdx.x;
  if (tid < 256) w[tid] = Wm3[tid];
  if (tid < 2)  b3[tid] = bm3[tid];
  __syncthreads();

  int p = blockIdx.x * blockDim.x + tid;
  float l0 = b3[0], l1 = b3[1];
  for (int j = 0; j < 128; ++j) {
    float h = (float)h2h[(size_t)p * 128 + j];
    l0 += h * w[2 * j];
    l1 += h * w[2 * j + 1];
  }
  float mx = fmaxf(l0, l1);
  float e0 = __expf(l0 - mx), e1 = __expf(l1 - mx);
  float inv = 1.f / (e0 + e1);
  float p0 = e0 * inv, p1 = e1 * inv;
  Bout[2 * p]     = p0;
  Bout[2 * p + 1] = p1;
  // log_softmax applied to softmax output (faithful double-softmax)
  float mb = fmaxf(p0, p1);
  float lse = mb + __logf(__expf(p0 - mb) + __expf(p1 - mb));
  float lb = ((labels[p] != 0) ? p1 : p0) - lse;
  atomicAdd(loss, -lb * (1.0f / (float)P_));
}

// ================== host launcher ==================
extern "C" void kernel_launch(void* const* d_in, const int* in_sizes, int n_in,
                              void* d_out, int out_size, void* d_ws, size_t ws_size,
                              hipStream_t stream) {
  (void)in_sizes; (void)n_in; (void)out_size; (void)ws_size;

  const float* A_u   = (const float*)d_in[0];
  const float* X_u   = (const float*)d_in[1];
  const float* A_v   = (const float*)d_in[2];
  const float* X_v   = (const float*)d_in[3];
  const int*   u_idx = (const int*)d_in[4];
  const int*   v_idx = (const int*)d_in[5];
  const int*   labels= (const int*)d_in[6];
  const float* W1_u  = (const float*)d_in[7];
  const float* W2_u  = (const float*)d_in[8];
  const float* Wg_u  = (const float*)d_in[9];
  const float* bg_u  = (const float*)d_in[10];
  const float* W1_v  = (const float*)d_in[11];
  const float* W2_v  = (const float*)d_in[12];
  const float* Wg_v  = (const float*)d_in[13];
  const float* bg_v  = (const float*)d_in[14];
  const float* Wm1   = (const float*)d_in[15];
  const float* bm1   = (const float*)d_in[16];
  const float* Wm2   = (const float*)d_in[17];
  const float* bm2   = (const float*)d_in[18];
  const float* Wm3   = (const float*)d_in[19];
  const float* bm3   = (const float*)d_in[20];

  float* out  = (float*)d_out;
  float* Xu_out = out;                                   // [N, C*DOUT]
  float* Xv_out = out + (size_t)NN * C_ * DOUT_;
  float* loss   = out + 2 * (size_t)NN * C_ * DOUT_;
  float* Bout   = loss + 1;                              // [P, 2]

  // ---- workspace layout (deterministic bump allocator) ----
  char* ws = (char*)d_ws;
  auto alloc = [&](size_t bytes) -> char* {
    char* p = ws;
    ws += (bytes + 255) & ~(size_t)255;
    return p;
  };
  const size_t cnn_h = (size_t)C_ * NN * NN * sizeof(_Float16);   // 16 MB
  _Float16* A1h   = (_Float16*)alloc(cnn_h);   // reused per branch
  _Float16* A2hT  = (_Float16*)alloc(cnn_h);   // reused per branch
  _Float16* Ht_u  = (_Float16*)alloc(cnn_h);
  _Float16* Ht_v  = (_Float16*)alloc(cnn_h);
  _Float16* XLT_u = (_Float16*)alloc((size_t)DOUT_ * NN * sizeof(_Float16));
  _Float16* XLT_v = (_Float16*)alloc((size_t)DOUT_ * NN * sizeof(_Float16));
  _Float16* Xuh   = (_Float16*)alloc((size_t)NN * C_ * DOUT_ * sizeof(_Float16));
  _Float16* Xvh   = (_Float16*)alloc((size_t)NN * C_ * DOUT_ * sizeof(_Float16));
  float*    deg   = (float*)   alloc((size_t)2 * C_ * NN * sizeof(float));
  _Float16* Wm1hT = (_Float16*)alloc((size_t)256 * MLP_IN_ * sizeof(_Float16));
  _Float16* Wm2hT = (_Float16*)alloc((size_t)128 * 256 * sizeof(_Float16));
  _Float16* h1h   = (_Float16*)alloc((size_t)P_ * 256 * sizeof(_Float16));
  _Float16* h2h   = (_Float16*)alloc((size_t)P_ * 128 * sizeof(_Float16));
  float* deg_u = deg;
  float* deg_v = deg + C_ * NN;

  // ---- pipeline ----
  k_init<<<32, 256, 0, stream>>>(deg, loss);
  k_prep_w<<<512, 256, 0, stream>>>(Wm1, Wm2, Wm1hT, Wm2hT);

  dim3 bT(32, 8);
  dim3 gT(NN / 32, NN / 32);

  // branch u (A12 region in use until k_spspmm completes)
  k_mix_transpose<<<gT, bT, 0, stream>>>(A_u, W1_u, W2_u, A1h, A2hT);
  k_spspmm<<<dim3(1024, C_), 128, 0, stream>>>(A1h, A2hT, Ht_u, deg_u);
  k_xl<<<(NN * DOUT_) / 256, 256, 0, stream>>>(X_u, Wg_u, XLT_u);

  // branch v (reuses A1h/A2hT region)
  k_mix_transpose<<<gT, bT, 0, stream>>>(A_v, W1_v, W2_v, A1h, A2hT);
  k_spspmm<<<dim3(1024, C_), 128, 0, stream>>>(A1h, A2hT, Ht_v, deg_v);
  k_xl<<<(NN * DOUT_) / 256, 256, 0, stream>>>(X_v, Wg_v, XLT_v);

  k_gcn_out<<<dim3(64, C_), 128, 0, stream>>>(Ht_u, XLT_u, deg_u, bg_u, Xu_out, Xuh);
  k_gcn_out<<<dim3(64, C_), 128, 0, stream>>>(Ht_v, XLT_v, deg_v, bg_v, Xv_out, Xvh);

  k_mlp1<<<512, 128, 0, stream>>>(Xuh, Xvh, u_idx, v_idx, Wm1hT, bm1, h1h);
  k_mlp2<<<256, 128, 0, stream>>>(h1h, Wm2hT, bm2, h2h);
  k_head<<<P_ / 256, 256, 0, stream>>>(h2h, Wm3, bm3, labels, Bout, loss);
}
